// Se3Diffusion_17420387353218
// MI455X (gfx1250) — compile-verified, matrix operand order checked by reading, not run on hardware
//
#include <hip/hip_runtime.h>
#include <math.h>

#define NN 1024
#define HIDN 256
#define KP 288      // 262 padded to multiple of 32
#define EH 128

typedef __bf16 v16bf __attribute__((ext_vector_type(16)));
typedef __bf16 v8bf  __attribute__((ext_vector_type(8)));
typedef float  v8f   __attribute__((ext_vector_type(8)));
typedef float  v2f   __attribute__((ext_vector_type(2)));

#if defined(__AMDGCN__) && __has_builtin(__builtin_amdgcn_wmma_f32_16x16x4_f32)
#define EDGE_WMMA_F32 1
#else
#define EDGE_WMMA_F32 0
#endif

union Frag { v16bf v; v8bf h[2]; };

__device__ __forceinline__ float silu_f(float x) { return x / (1.0f + __expf(-x)); }

__device__ __forceinline__ void norm3(float& a, float& b, float& c) {
  float ss = a*a + b*b + c*c;
  float inv = 1.0f / sqrtf(fmaxf(ss, 1e-8f));
  a *= inv; b *= inv; c *= inv;
}

// ---------------------------------------------------------------------------
// Per-node prep: frames, delta_local, x_current, padded bf16 node_feat row
// ---------------------------------------------------------------------------
__global__ void k_prep_node(const float* __restrict__ h, const float* __restrict__ dn,
                            const float* __restrict__ xc, const float* __restrict__ ts,
                            float* __restrict__ frames, float* __restrict__ dl,
                            float* __restrict__ xcur, __bf16* __restrict__ nf) {
  int i = blockIdx.x * blockDim.x + threadIdx.x;
  if (i >= NN) return;
  float x0[3], pv[3], nx[3];
  #pragma unroll
  for (int c = 0; c < 3; c++) x0[c] = xc[i*3 + c];
  #pragma unroll
  for (int c = 0; c < 3; c++) {
    pv[c] = (i == 0)      ? (xc[3 + c] - xc[c])                       : (x0[c] - xc[(i-1)*3 + c]);
    nx[c] = (i == NN - 1) ? (xc[(NN-1)*3 + c] - xc[(NN-2)*3 + c])     : (xc[(i+1)*3 + c] - x0[c]);
  }
  float t0 = pv[0]+nx[0], t1 = pv[1]+nx[1], t2 = pv[2]+nx[2]; norm3(t0, t1, t2);
  float c0 = nx[0]-pv[0], c1 = nx[1]-pv[1], c2 = nx[2]-pv[2]; norm3(c0, c1, c2);
  float dt = c0*t0 + c1*t1 + c2*t2;
  float ay0 = c0 - t0*dt, ay1 = c1 - t1*dt, ay2 = c2 - t2*dt; norm3(ay0, ay1, ay2);
  float az0 = t1*ay2 - t2*ay1, az1 = t2*ay0 - t0*ay2, az2 = t0*ay1 - t1*ay0; norm3(az0, az1, az2);
  ay0 = az1*t2 - az2*t1; ay1 = az2*t0 - az0*t2; ay2 = az0*t1 - az1*t0; norm3(ay0, ay1, ay2);
  // frames[j][i2] row-major: row j = xyz component, col = (tangent, ay, az)
  float F[9] = { t0, ay0, az0,  t1, ay1, az1,  t2, ay2, az2 };
  #pragma unroll
  for (int q = 0; q < 9; q++) frames[i*9 + q] = F[q];
  float d0 = dn[i*3], d1 = dn[i*3+1], d2 = dn[i*3+2];
  float dl0 = F[0]*d0 + F[3]*d1 + F[6]*d2;
  float dl1 = F[1]*d0 + F[4]*d1 + F[7]*d2;
  float dl2 = F[2]*d0 + F[5]*d1 + F[8]*d2;
  dl[i*3] = dl0; dl[i*3+1] = dl1; dl[i*3+2] = dl2;
  #pragma unroll
  for (int c = 0; c < 3; c++) xcur[i*3 + c] = x0[c] + dn[i*3 + c];
  float tv = ts[0];
  __bf16* row = nf + (size_t)i * KP;
  for (int c = 0; c < HIDN; c++) row[c] = (__bf16)h[i*HIDN + c];
  row[256] = (__bf16)dl0; row[257] = (__bf16)dl1; row[258] = (__bf16)dl2;
  row[259] = (__bf16)tv;  row[260] = (__bf16)sinf(tv); row[261] = (__bf16)cosf(tv);
  for (int c = 262; c < KP; c++) row[c] = (__bf16)0.0f;
}

// ---------------------------------------------------------------------------
// Transpose + pad weights to bf16 [out][K] so WMMA B frags are contiguous
// ---------------------------------------------------------------------------
__global__ void k_prep_w(const float* __restrict__ W1, const float* __restrict__ W2,
                         const float* __restrict__ Wo1,
                         __bf16* __restrict__ W1t, __bf16* __restrict__ W2t,
                         __bf16* __restrict__ Wo1t) {
  int idx = blockIdx.x * blockDim.x + threadIdx.x;
  if (idx < 256 * KP) {
    int o = idx / KP, k = idx % KP;
    float a = (k < 262) ? W1[k*256 + o]  : 0.0f;
    float b = (k < 262) ? Wo1[k*256 + o] : 0.0f;
    W1t[idx]  = (__bf16)a;
    Wo1t[idx] = (__bf16)b;
  }
  int idx2 = idx - 256 * KP;
  if (idx2 >= 0 && idx2 < 256 * 256) {
    int o = idx2 / 256, k = idx2 % 256;
    W2t[idx2] = (__bf16)W2[k*256 + o];
  }
}

// ---------------------------------------------------------------------------
// Generic bf16 WMMA GEMM: C = silu(A[MxK] * Bt[NxK]^T + bias), wave = 16x16 tile
// ---------------------------------------------------------------------------
__global__ void k_gemm(const __bf16* __restrict__ A, const __bf16* __restrict__ Bt,
                       const float* __restrict__ bias, float* __restrict__ Cf,
                       __bf16* __restrict__ Cb, int M, int Ncols, int K) {
  int wid  = (blockIdx.x * blockDim.x + threadIdx.x) >> 5;
  int lane = threadIdx.x & 31;
  int tn_cnt = Ncols >> 4;
  int tm = wid / tn_cnt;
  int tn = wid - tm * tn_cnt;
  if (tm * 16 >= M) return;
  int lm = lane & 15;
  int hh = (lane >> 4) * 8;                 // K-half offset within 16
  const __bf16* arow = A  + (size_t)(tm*16 + lm) * K;
  const __bf16* brow = Bt + (size_t)(tn*16 + lm) * K;
  v8f acc = {};
  for (int kc = 0; kc < K; kc += 32) {
    Frag a, b;
    a.h[0] = *(const v8bf*)(arow + kc + hh);
    a.h[1] = *(const v8bf*)(arow + kc + 16 + hh);
    b.h[0] = *(const v8bf*)(brow + kc + hh);
    b.h[1] = *(const v8bf*)(brow + kc + 16 + hh);
    acc = __builtin_amdgcn_wmma_f32_16x16x32_bf16(false, a.v, false, b.v,
                                                  (short)0, acc, false, false);
  }
  int nc = tn*16 + lm;
  float bv = bias[nc];
  int mbase = tm*16 + hh;                   // D rows: r + 8*(lane>>4)
  #pragma unroll
  for (int r = 0; r < 8; r++) {
    float v = silu_f(acc[r] + bv);
    size_t off = (size_t)(mbase + r) * Ncols + nc;
    if (Cf) Cf[off] = v;
    if (Cb) Cb[off] = (__bf16)v;
  }
}

// ---------------------------------------------------------------------------
// Per-node scalar heads: src logit, dst logit, sigmoid(ns)
// ---------------------------------------------------------------------------
__global__ void k_node_scalars(const float* __restrict__ nh,
                               const float* __restrict__ wsrc, const float* __restrict__ bsrc,
                               const float* __restrict__ wdst, const float* __restrict__ bdst,
                               const float* __restrict__ wns,  const float* __restrict__ bns,
                               float* __restrict__ nscal) {
  int i = blockIdx.x * blockDim.x + threadIdx.x;
  if (i >= NN) return;
  float s = 0.f, d = 0.f, ns = 0.f;
  for (int k = 0; k < HIDN; k++) {
    float v = nh[i*HIDN + k];
    s  += v * wsrc[k];
    d  += v * wdst[k];
    ns += v * wns[k];
  }
  nscal[i*3 + 0] = s + bsrc[0];
  nscal[i*3 + 1] = d + bdst[0];
  nscal[i*3 + 2] = 1.0f / (1.0f + __expf(-(ns + bns[0])));
}

// ---------------------------------------------------------------------------
// Fused edge MLP -> logits. One wave = (row n, 16 columns m).
// Layer1: exact-shape V_WMMA_F32_16X16X4_F32 (K=4, all lanes live) when the
// builtin lowers; otherwise K-padded bf16 16x16x32. silu + We2 dot in-register.
// ---------------------------------------------------------------------------
__global__ void k_edge(const float* __restrict__ frames, const float* __restrict__ xcur,
                       const float* __restrict__ nscal,
                       const float* __restrict__ We1, const float* __restrict__ be1,
                       const float* __restrict__ We2, const float* __restrict__ be2,
                       float* __restrict__ logits) {
  int gw   = blockIdx.x * (blockDim.x >> 5) + (threadIdx.x >> 5);
  int lane = threadIdx.x & 31;
  int n  = gw >> 6;                   // 64 m-chunks of 16 per row
  int mc = gw & 63;
  if (n >= NN) return;
  int lm = lane & 15;
  int half = lane >> 4;
  int m  = mc*16 + lm;

  float F[9];
  #pragma unroll
  for (int q = 0; q < 9; q++) F[q] = frames[n*9 + q];
  float xn0 = xcur[n*3], xn1 = xcur[n*3+1], xn2 = xcur[n*3+2];
  float r0 = xcur[m*3]   - xn0;
  float r1 = xcur[m*3+1] - xn1;
  float r2 = xcur[m*3+2] - xn2;
  float rl0 = F[0]*r0 + F[3]*r1 + F[6]*r2;
  float rl1 = F[1]*r0 + F[4]*r1 + F[7]*r2;
  float rl2 = F[2]*r0 + F[5]*r1 + F[8]*r2;
  float dist = sqrtf(rl0*rl0 + rl1*rl1 + rl2*rl2);

  float part[8] = {0.f, 0.f, 0.f, 0.f, 0.f, 0.f, 0.f, 0.f};

#if EDGE_WMMA_F32
  // A 16x4 f32: lanes 0-15 hold {K0,K1} of row lm; lanes 16-31 hold {K2,K3}.
  v2f a;
  a.x = half ? rl2  : rl0;
  a.y = half ? dist : rl1;
  for (int c = 0; c < 8; c++) {              // 8 chunks of 16 hidden units
    int col = c*16 + lm;
    v2f b;                                   // B 4x16: col per lane, K split by half
    b.x = We1[(half ? 2 : 0)*EH + col];
    b.y = We1[(half ? 3 : 1)*EH + col];
    v8f dz = {};
    dz = __builtin_amdgcn_wmma_f32_16x16x4_f32(false, a, false, b,
                                               (short)0, dz, false, false);
    float w2 = We2[col];
    float bb = be1[col];
    #pragma unroll
    for (int r = 0; r < 8; r++)
      part[r] += silu_f(dz[r] + bb) * w2;    // lane owns hidden unit `col`, row r+8*half
  }
#else
  // Fallback: K=4 zero-padded into bf16 16x16x32. Lanes 0-15 hold K=0..7.
  Frag a;
  #pragma unroll
  for (int q = 0; q < 16; q++) a.v[q] = (__bf16)0.0f;
  if (lane < 16) {
    a.v[0] = (__bf16)rl0; a.v[1] = (__bf16)rl1;
    a.v[2] = (__bf16)rl2; a.v[3] = (__bf16)dist;
  }
  for (int c = 0; c < 8; c++) {
    int col = c*16 + lm;
    Frag b;
    #pragma unroll
    for (int q = 0; q < 16; q++) b.v[q] = (__bf16)0.0f;
    if (lane < 16) {
      #pragma unroll
      for (int k = 0; k < 4; k++) b.v[k] = (__bf16)We1[k*EH + col];
    }
    v8f dz = {};
    dz = __builtin_amdgcn_wmma_f32_16x16x32_bf16(false, a.v, false, b.v,
                                                 (short)0, dz, false, false);
    float w2 = We2[col];
    float bb = be1[col];
    #pragma unroll
    for (int r = 0; r < 8; r++)
      part[r] += silu_f(dz[r] + bb) * w2;
  }
#endif

  // Sum over the 16 hidden-unit lanes in each half-wave
  #pragma unroll
  for (int off = 1; off < 16; off <<= 1) {
    #pragma unroll
    for (int r = 0; r < 8; r++) part[r] += __shfl_xor(part[r], off, 32);
  }
  if (lm == 0) {
    float srcn = nscal[n*3 + 0];
    int base = half * 8;
    for (int r = 0; r < 8; r++) {
      int mr = mc*16 + base + r;
      float lg = part[r] + be2[0] + srcn + nscal[mr*3 + 1];
      if (mr == n) lg = -10000.0f;
      logits[(size_t)n*NN + mr] = lg;
    }
  }
}

// ---------------------------------------------------------------------------
// Row softmax (diag masked + renorm) and attention-weighted local messages
// ---------------------------------------------------------------------------
__global__ void k_softmax_msg(const float* __restrict__ logits, const float* __restrict__ frames,
                              const float* __restrict__ xcur, const float* __restrict__ nscal,
                              float* __restrict__ msg) {
  __shared__ float red[256];
  int n = blockIdx.x, t = threadIdx.x;
  const float* lrow = logits + (size_t)n * NN;

  float mx = -1e30f;
  for (int m = t; m < NN; m += 256) mx = fmaxf(mx, lrow[m]);
  red[t] = mx; __syncthreads();
  for (int s = 128; s > 0; s >>= 1) { if (t < s) red[t] = fmaxf(red[t], red[t+s]); __syncthreads(); }
  mx = red[0]; __syncthreads();

  float sm = 0.f;
  for (int m = t; m < NN; m += 256) if (m != n) sm += __expf(lrow[m] - mx);
  red[t] = sm; __syncthreads();
  for (int s = 128; s > 0; s >>= 1) { if (t < s) red[t] += red[t+s]; __syncthreads(); }
  float inv = 1.0f / fmaxf(red[0], 1e-6f);
  __syncthreads();

  float F[9];
  #pragma unroll
  for (int q = 0; q < 9; q++) F[q] = frames[n*9 + q];
  float xn0 = xcur[n*3], xn1 = xcur[n*3+1], xn2 = xcur[n*3+2];
  float a0 = 0.f, a1 = 0.f, a2 = 0.f;
  for (int m = t; m < NN; m += 256) if (m != n) {
    float w = __expf(lrow[m] - mx) * inv * nscal[m*3 + 2];
    float r0 = xcur[m*3]   - xn0;
    float r1 = xcur[m*3+1] - xn1;
    float r2 = xcur[m*3+2] - xn2;
    a0 += w * (F[0]*r0 + F[3]*r1 + F[6]*r2);
    a1 += w * (F[1]*r0 + F[4]*r1 + F[7]*r2);
    a2 += w * (F[2]*r0 + F[5]*r1 + F[8]*r2);
  }
  red[t] = a0; __syncthreads();
  for (int s = 128; s > 0; s >>= 1) { if (t < s) red[t] += red[t+s]; __syncthreads(); }
  if (t == 0) msg[n*3 + 0] = red[0];
  __syncthreads();
  red[t] = a1; __syncthreads();
  for (int s = 128; s > 0; s >>= 1) { if (t < s) red[t] += red[t+s]; __syncthreads(); }
  if (t == 0) msg[n*3 + 1] = red[0];
  __syncthreads();
  red[t] = a2; __syncthreads();
  for (int s = 128; s > 0; s >>= 1) { if (t < s) red[t] += red[t+s]; __syncthreads(); }
  if (t == 0) msg[n*3 + 2] = red[0];
}

// ---------------------------------------------------------------------------
// Build bf16 update_feat = [node_hidden(256) | delta_local(3) | msg(3) | pad]
// ---------------------------------------------------------------------------
__global__ void k_build_update(const float* __restrict__ nh, const float* __restrict__ dl,
                               const float* __restrict__ msg, __bf16* __restrict__ uf) {
  int idx = blockIdx.x * blockDim.x + threadIdx.x;
  if (idx >= NN * KP) return;
  int i = idx / KP, c = idx % KP;
  float v = 0.f;
  if (c < 256)      v = nh[i*HIDN + c];
  else if (c < 259) v = dl[i*3 + (c - 256)];
  else if (c < 262) v = msg[i*3 + (c - 259)];
  uf[idx] = (__bf16)v;
}

// ---------------------------------------------------------------------------
// Output head: noise_local = h3 @ Wo2 + bo2 + 0.25*msg ; rotate back by frame
// ---------------------------------------------------------------------------
__global__ void k_final(const float* __restrict__ h3, const float* __restrict__ Wo2,
                        const float* __restrict__ bo2, const float* __restrict__ msg,
                        const float* __restrict__ frames, float* __restrict__ out) {
  int i = blockIdx.x * blockDim.x + threadIdx.x;
  if (i >= NN) return;
  float nl[3] = { bo2[0], bo2[1], bo2[2] };
  for (int k = 0; k < HIDN; k++) {
    float v = h3[i*HIDN + k];
    nl[0] += v * Wo2[k*3 + 0];
    nl[1] += v * Wo2[k*3 + 1];
    nl[2] += v * Wo2[k*3 + 2];
  }
  #pragma unroll
  for (int c = 0; c < 3; c++) nl[c] += 0.25f * msg[i*3 + c];
  const float* F = frames + i*9;
  out[i*3 + 0] = F[0]*nl[0] + F[1]*nl[1] + F[2]*nl[2];
  out[i*3 + 1] = F[3]*nl[0] + F[4]*nl[1] + F[5]*nl[2];
  out[i*3 + 2] = F[6]*nl[0] + F[7]*nl[1] + F[8]*nl[2];
}

extern "C" void kernel_launch(void* const* d_in, const int* in_sizes, int n_in,
                              void* d_out, int out_size, void* d_ws, size_t ws_size,
                              hipStream_t stream) {
  const float* h    = (const float*)d_in[0];
  const float* dn   = (const float*)d_in[1];
  const float* xc   = (const float*)d_in[2];
  const float* ts   = (const float*)d_in[3];
  const float* W1   = (const float*)d_in[4];
  const float* b1   = (const float*)d_in[5];
  const float* W2   = (const float*)d_in[6];
  const float* b2   = (const float*)d_in[7];
  const float* wsrc = (const float*)d_in[8];
  const float* bsrc = (const float*)d_in[9];
  const float* wdst = (const float*)d_in[10];
  const float* bdst = (const float*)d_in[11];
  const float* wns  = (const float*)d_in[12];
  const float* bns  = (const float*)d_in[13];
  const float* We1  = (const float*)d_in[14];
  const float* be1  = (const float*)d_in[15];
  const float* We2  = (const float*)d_in[16];
  const float* be2  = (const float*)d_in[17];
  const float* Wo1  = (const float*)d_in[18];
  const float* bo1  = (const float*)d_in[19];
  const float* Wo2  = (const float*)d_in[20];
  const float* bo2  = (const float*)d_in[21];
  float* out = (float*)d_out;

  char* p = (char*)d_ws;
  auto alloc = [&](size_t bytes) -> void* {
    void* r = (void*)p;
    p += (bytes + 255) & ~(size_t)255;
    return r;
  };
  float*  frames = (float*) alloc((size_t)NN * 9 * 4);
  float*  dl     = (float*) alloc((size_t)NN * 3 * 4);
  float*  xcur   = (float*) alloc((size_t)NN * 3 * 4);
  __bf16* nf     = (__bf16*)alloc((size_t)NN * KP * 2);
  __bf16* W1t    = (__bf16*)alloc((size_t)256 * KP * 2);
  __bf16* W2t    = (__bf16*)alloc((size_t)256 * 256 * 2);
  __bf16* Wo1t   = (__bf16*)alloc((size_t)256 * KP * 2);
  __bf16* h1b    = (__bf16*)alloc((size_t)NN * HIDN * 2);
  float*  nh     = (float*) alloc((size_t)NN * HIDN * 4);
  float*  nscal  = (float*) alloc((size_t)NN * 3 * 4);
  float*  logits = (float*) alloc((size_t)NN * NN * 4);
  float*  msg    = (float*) alloc((size_t)NN * 3 * 4);
  __bf16* uf     = (__bf16*)alloc((size_t)NN * KP * 2);
  float*  h3     = (float*) alloc((size_t)NN * HIDN * 4);

  k_prep_node<<<NN/256, 256, 0, stream>>>(h, dn, xc, ts, frames, dl, xcur, nf);
  k_prep_w<<<(256*KP + 256*256 + 255)/256, 256, 0, stream>>>(W1, W2, Wo1, W1t, W2t, Wo1t);

  // node MLP: 1024x288 @ 288x256 -> silu -> (bf16), then @256x256 -> silu -> f32
  k_gemm<<<(NN/16)*(HIDN/16)*32/256, 256, 0, stream>>>(nf, W1t, b1, nullptr, h1b, NN, HIDN, KP);
  k_gemm<<<(NN/16)*(HIDN/16)*32/256, 256, 0, stream>>>(h1b, W2t, b2, nh, nullptr, NN, HIDN, HIDN);

  k_node_scalars<<<NN/256, 256, 0, stream>>>(nh, wsrc, bsrc, wdst, bdst, wns, bns, nscal);

  // fused edge MLP: 1024 rows x 64 chunks = 65536 waves, 8 waves/block
  k_edge<<<NN*64/8, 256, 0, stream>>>(frames, xcur, nscal, We1, be1, We2, be2, logits);

  k_softmax_msg<<<NN, 256, 0, stream>>>(logits, frames, xcur, nscal, msg);

  k_build_update<<<(NN*KP)/256, 256, 0, stream>>>(nh, dl, msg, uf);
  k_gemm<<<(NN/16)*(HIDN/16)*32/256, 256, 0, stream>>>(uf, Wo1t, bo1, h3, nullptr, NN, HIDN, KP);

  k_final<<<NN/128, 128, 0, stream>>>(h3, Wo2, bo2, msg, frames, out);
}
